// MoELayer_8504035246348
// MI455X (gfx1250) — matrix operand order repaired and checked by reading, not compiled, not run on hardware
//
#include <hip/hip_runtime.h>
#include <hip/hip_bf16.h>
#include <math.h>

// ---------------------------------------------------------------------------
// MoE layer for MI455X (gfx1250): wave32, WMMA bf16 16x16x32, fp32 accumulate,
// async LDS copies (ASYNCcnt) + prefetch for latency hiding.
// ---------------------------------------------------------------------------

typedef __attribute__((ext_vector_type(16))) __bf16 v16bf;
typedef __attribute__((ext_vector_type(8)))  float  v8f;

#define T_TOK 8192
#define DDIM  1024
#define FDIM  2048
#define NEXP  8

// LDS row strides (in bf16 elements). 40 elems = 80B = 20 banks -> conflict-free.
#define LDA 40
#define LDBS 40

__device__ __forceinline__ v8f wmma_bf16(v16bf a, v16bf b, v8f c) {
  // (neg_a, A, neg_b, B, c_mod, C, reuse_a, reuse_b)
  return __builtin_amdgcn_wmma_f32_16x16x32_bf16(false, a, false, b, (short)0, c,
                                                 false, false);
}

// A fragment (16x32 bf16, M x K): lane m = base + (lane&15);
// 16 values: K = (lane>=16)*8 + {0..7}, then +16  => two contiguous 16B chunks.
__device__ __forceinline__ v16bf frag_a(const __bf16* lds, int mbase, int lane) {
  int m  = mbase + (lane & 15);
  int kb = (lane >> 4) * 8;
  union { v16bf v; uint4 q[2]; } u;
  u.q[0] = *reinterpret_cast<const uint4*>(lds + m * LDA + kb);
  u.q[1] = *reinterpret_cast<const uint4*>(lds + m * LDA + kb + 16);
  return u.v;
}

// B fragment (32x16 bf16, K x N), staged transposed in LDS as [N][K]:
// lane n = base + (lane&15); 16 values K = (lane>=16)*8 + {0..7}, then +16.
__device__ __forceinline__ v16bf frag_b(const __bf16* lds, int nbase, int lane) {
  int n  = nbase + (lane & 15);
  int kb = (lane >> 4) * 8;
  union { v16bf v; uint4 q[2]; } u;
  u.q[0] = *reinterpret_cast<const uint4*>(lds + n * LDBS + kb);
  u.q[1] = *reinterpret_cast<const uint4*>(lds + n * LDBS + kb + 16);
  return u.v;
}

// Async global -> LDS 16B copy (tracked by ASYNCcnt, no VGPR round trip).
__device__ __forceinline__ void async_copy_b128(unsigned lds_addr,
                                                unsigned long long gaddr) {
  asm volatile("global_load_async_to_lds_b128 %0, %1, off"
               :: "v"(lds_addr), "v"(gaddr)
               : "memory");
}
__device__ __forceinline__ void wait_asynccnt0() {
  asm volatile("s_wait_asynccnt 0" ::: "memory");
}

// ---------------------------------------------------------------------------
// Router: one wave (32 lanes) per token. Full fp32 softmax/top-2/renorm.
// ---------------------------------------------------------------------------
__global__ __launch_bounds__(256)
void router_kernel(const float* __restrict__ x, const float* __restrict__ gw,
                   const float* __restrict__ sgw, float* __restrict__ cw,
                   float* __restrict__ sgate) {
  int wave = threadIdx.x >> 5, lane = threadIdx.x & 31;
  int t = blockIdx.x * 8 + wave;
  float acc0 = 0.f, acc1 = 0.f, acc2 = 0.f, acc3 = 0.f;
  float acc4 = 0.f, acc5 = 0.f, acc6 = 0.f, acc7 = 0.f, sacc = 0.f;
  for (int d0 = 0; d0 < DDIM; d0 += 32) {
    int d = d0 + lane;
    float xv = x[(size_t)t * DDIM + d];
    const float4* g = reinterpret_cast<const float4*>(gw + (size_t)d * 8);
    float4 ga = g[0], gb = g[1];
    acc0 += xv * ga.x; acc1 += xv * ga.y; acc2 += xv * ga.z; acc3 += xv * ga.w;
    acc4 += xv * gb.x; acc5 += xv * gb.y; acc6 += xv * gb.z; acc7 += xv * gb.w;
    sacc += xv * sgw[d];
  }
  for (int off = 16; off > 0; off >>= 1) {  // wave32 butterfly
    acc0 += __shfl_xor(acc0, off); acc1 += __shfl_xor(acc1, off);
    acc2 += __shfl_xor(acc2, off); acc3 += __shfl_xor(acc3, off);
    acc4 += __shfl_xor(acc4, off); acc5 += __shfl_xor(acc5, off);
    acc6 += __shfl_xor(acc6, off); acc7 += __shfl_xor(acc7, off);
    sacc += __shfl_xor(sacc, off);
  }
  if (lane == 0) {
    float l[8] = {acc0, acc1, acc2, acc3, acc4, acc5, acc6, acc7};
    int i0 = 0; float m0 = l[0];
    for (int e = 1; e < 8; ++e) if (l[e] > m0) { m0 = l[e]; i0 = e; }
    int i1 = -1; float m1 = -3.4e38f;
    for (int e = 0; e < 8; ++e) if (e != i0 && l[e] > m1) { m1 = l[e]; i1 = e; }
    float w0 = 1.f / (1.f + __expf(m1 - m0));  // p0/(p0+p1), stable
    float w1 = 1.f - w0;
    float row[8];
    for (int e = 0; e < 8; ++e) row[e] = 0.f;
    row[i0] = w0; row[i1] = w1;
    for (int e = 0; e < 8; ++e) cw[(size_t)t * 8 + e] = row[e];
    sgate[t] = 1.f / (1.f + __expf(-sacc));
  }
}

// ---------------------------------------------------------------------------
// Scatter: build compacted per-expert token lists (routed-only compute: 2/8).
// ---------------------------------------------------------------------------
__global__ void scatter_kernel(const float* __restrict__ cw, int* __restrict__ cnt,
                               int* __restrict__ idx) {
  int t = blockIdx.x * blockDim.x + threadIdx.x;
  if (t >= T_TOK) return;
  for (int e = 0; e < NEXP; ++e) {
    if (cw[(size_t)t * 8 + e] > 0.f) {
      int pos = atomicAdd(&cnt[e], 1);
      idx[e * T_TOK + pos] = t;
    }
  }
}

// ---------------------------------------------------------------------------
// Fused dual GEMM: h = silu(Xg @ W1) * (Xg @ W3), Xg = gathered token rows.
// Block tile 128(M) x 64(N), 8 waves as 4x2 of 32x32, K chunk 32.
// Register-pipelined: global loads for chunk k+1 issued before WMMAs of k.
// ---------------------------------------------------------------------------
__global__ __launch_bounds__(256)
void gemm13_kernel(const float* __restrict__ X,
                   const float* __restrict__ W1, const float* __restrict__ W3,
                   const int* __restrict__ idxList, const int* __restrict__ countPtr,
                   int fixedCount, __bf16* __restrict__ hout) {
  __shared__ int tokLDS[128];
  __shared__ __align__(16) __bf16 ldsA[128 * LDA];
  __shared__ __align__(16) __bf16 ldsB1[64 * LDBS];
  __shared__ __align__(16) __bf16 ldsB3[64 * LDBS];

  int count = countPtr ? *countPtr : fixedCount;
  int mtile = blockIdx.y;
  if (mtile * 128 >= count) return;  // dynamic-M early exit
  int n0 = blockIdx.x * 64;

  int tid = threadIdx.x;
  int wave = tid >> 5, lane = tid & 31;
  int waveM = wave >> 1, waveN = wave & 1;

  if (tid < 128) {
    int m = mtile * 128 + tid;
    tokLDS[tid] = (m < count) ? (idxList ? idxList[m] : m) : -1;
  }
  __syncthreads();

  // Cache the 4 gather tokens this thread stages (rows tid>>3 + i*32).
  int tokR[4];
#pragma unroll
  for (int i = 0; i < 4; ++i) tokR[i] = tokLDS[(tid >> 3) + i * 32];

  v8f z = {0.f, 0.f, 0.f, 0.f, 0.f, 0.f, 0.f, 0.f};
  v8f acc1[2][2] = {{z, z}, {z, z}};
  v8f acc3[2][2] = {{z, z}, {z, z}};

  float4 aR[4], b1R[2], b3R[2];

  auto load_chunk = [&](int k0) {
#pragma unroll
    for (int i = 0; i < 4; ++i) {          // A: 128x32 fp32 gathered rows
      int lin4 = tid + i * 256;
      int c4 = (lin4 & 7) * 4;
      int tok = tokR[i];
      aR[i] = make_float4(0.f, 0.f, 0.f, 0.f);
      if (tok >= 0)
        aR[i] = *reinterpret_cast<const float4*>(X + (size_t)tok * DDIM + k0 + c4);
    }
#pragma unroll
    for (int i = 0; i < 2; ++i) {          // B1/B3: 32(K) x 64(N) fp32
      int lin4 = tid + i * 256;
      int kk = lin4 >> 4;
      int n4 = (lin4 & 15) * 4;
      size_t goff = (size_t)(k0 + kk) * FDIM + n0 + n4;
      b1R[i] = *reinterpret_cast<const float4*>(W1 + goff);
      b3R[i] = *reinterpret_cast<const float4*>(W3 + goff);
      if (k0 + 32 < DDIM) {                // prefetch chunk after next
        __builtin_prefetch(W1 + goff + (size_t)32 * FDIM, 0, 1);
        __builtin_prefetch(W3 + goff + (size_t)32 * FDIM, 0, 1);
      }
    }
  };

  auto store_chunk = [&]() {
#pragma unroll
    for (int i = 0; i < 4; ++i) {          // A -> bf16 LDS [M][K]
      int lin4 = tid + i * 256;
      int row = lin4 >> 3;
      int c4 = (lin4 & 7) * 4;
      union { __bf16 h[4]; uint2 u; } cvt;
      cvt.h[0] = (__bf16)aR[i].x; cvt.h[1] = (__bf16)aR[i].y;
      cvt.h[2] = (__bf16)aR[i].z; cvt.h[3] = (__bf16)aR[i].w;
      *reinterpret_cast<uint2*>(&ldsA[row * LDA + c4]) = cvt.u;
    }
#pragma unroll
    for (int i = 0; i < 2; ++i) {          // B -> transposed bf16 LDS [N][K]
      int lin4 = tid + i * 256;
      int kk = lin4 >> 4;
      int n4 = (lin4 & 15) * 4;
      ldsB1[(n4 + 0) * LDBS + kk] = (__bf16)b1R[i].x;
      ldsB1[(n4 + 1) * LDBS + kk] = (__bf16)b1R[i].y;
      ldsB1[(n4 + 2) * LDBS + kk] = (__bf16)b1R[i].z;
      ldsB1[(n4 + 3) * LDBS + kk] = (__bf16)b1R[i].w;
      ldsB3[(n4 + 0) * LDBS + kk] = (__bf16)b3R[i].x;
      ldsB3[(n4 + 1) * LDBS + kk] = (__bf16)b3R[i].y;
      ldsB3[(n4 + 2) * LDBS + kk] = (__bf16)b3R[i].z;
      ldsB3[(n4 + 3) * LDBS + kk] = (__bf16)b3R[i].w;
    }
  };

  load_chunk(0);
  for (int k0 = 0; k0 < DDIM; k0 += 32) {
    store_chunk();
    __syncthreads();
    if (k0 + 32 < DDIM) load_chunk(k0 + 32);  // overlaps with WMMAs below

    v16bf a0 = frag_a(ldsA, waveM * 32 + 0, lane);
    v16bf a1 = frag_a(ldsA, waveM * 32 + 16, lane);
#pragma unroll
    for (int ns = 0; ns < 2; ++ns) {
      v16bf b1 = frag_b(ldsB1, waveN * 32 + ns * 16, lane);
      v16bf b3 = frag_b(ldsB3, waveN * 32 + ns * 16, lane);
      acc1[0][ns] = wmma_bf16(a0, b1, acc1[0][ns]);
      acc1[1][ns] = wmma_bf16(a1, b1, acc1[1][ns]);
      acc3[0][ns] = wmma_bf16(a0, b3, acc3[0][ns]);
      acc3[1][ns] = wmma_bf16(a1, b3, acc3[1][ns]);
    }
    __syncthreads();
  }

  // Epilogue: h = silu(acc1) * acc3 -> bf16. C layout: reg r -> M = r + 8*(lane>=16)
#pragma unroll
  for (int ms = 0; ms < 2; ++ms) {
#pragma unroll
    for (int ns = 0; ns < 2; ++ns) {
      int mbase = mtile * 128 + waveM * 32 + ms * 16 + (lane >> 4) * 8;
      int n = n0 + waveN * 32 + ns * 16 + (lane & 15);
#pragma unroll
      for (int r = 0; r < 8; ++r) {
        int m = mbase + r;
        if (m < count) {
          float v1 = acc1[ms][ns][r];
          float v3 = acc3[ms][ns][r];
          float hv = (v1 / (1.f + __expf(-v1))) * v3;  // silu(v1) * v3
          hout[(size_t)m * FDIM + n] = (__bf16)hv;
        }
      }
    }
  }
}

// ---------------------------------------------------------------------------
// GEMM2: out[token, :] (+)= scale[token] * (h @ W2).
// A (bf16 h) staged via GLOBAL_LOAD_ASYNC_TO_LDS_B128 into double-buffered LDS
// (ASYNCcnt fenced); B (fp32 w2) register-pipelined with inline bf16 convert.
// ---------------------------------------------------------------------------
__global__ __launch_bounds__(256)
void gemm2_kernel(const __bf16* __restrict__ H, const float* __restrict__ W2,
                  const int* __restrict__ idxList, const int* __restrict__ countPtr,
                  int fixedCount, const float* __restrict__ scale, int scaleStride,
                  float* __restrict__ out, int atomicMode) {
  __shared__ int   tokLDS[128];
  __shared__ float sLDS[128];
  __shared__ __align__(16) __bf16 ldsA[2][128 * LDA];
  __shared__ __align__(16) __bf16 ldsB[64 * LDBS];

  int count = countPtr ? *countPtr : fixedCount;
  int mtile = blockIdx.y;
  if (mtile * 128 >= count) return;
  int n0 = blockIdx.x * 64;

  int tid = threadIdx.x;
  int wave = tid >> 5, lane = tid & 31;
  int waveM = wave >> 1, waveN = wave & 1;

  if (tid < 128) {
    int m = mtile * 128 + tid;
    if (m < count) {
      int tok = idxList ? idxList[m] : m;
      tokLDS[tid] = tok;
      sLDS[tid] = scale[(size_t)tok * scaleStride];
    } else {
      tokLDS[tid] = 0;
      sLDS[tid] = 0.f;
    }
  }

  v8f z = {0.f, 0.f, 0.f, 0.f, 0.f, 0.f, 0.f, 0.f};
  v8f acc[2][2] = {{z, z}, {z, z}};

  float4 bR[2];

  // Async-copy A chunk (128x32 bf16 = 512 x 16B) into LDS buffer p.
  auto async_a = [&](int k0, int p) {
#pragma unroll
    for (int i = 0; i < 2; ++i) {
      int lin8 = tid + i * 256;
      int row = lin8 >> 2;
      int c8 = (lin8 & 3) * 8;
      const __bf16* g = H + (size_t)(mtile * 128 + row) * FDIM + k0 + c8;
      unsigned laddr =
          (unsigned)(unsigned long long)(&ldsA[p][row * LDA + c8]);
      async_copy_b128(laddr, (unsigned long long)g);
    }
  };
  auto load_b = [&](int k0) {
#pragma unroll
    for (int i = 0; i < 2; ++i) {
      int lin4 = tid + i * 256;
      int kk = lin4 >> 4;
      int n4 = (lin4 & 15) * 4;
      size_t goff = (size_t)(k0 + kk) * DDIM + n0 + n4;
      bR[i] = *reinterpret_cast<const float4*>(W2 + goff);
      if (k0 + 32 < FDIM)
        __builtin_prefetch(W2 + goff + (size_t)32 * DDIM, 0, 1);
    }
  };
  auto store_b = [&]() {
#pragma unroll
    for (int i = 0; i < 2; ++i) {
      int lin4 = tid + i * 256;
      int kk = lin4 >> 4;
      int n4 = (lin4 & 15) * 4;
      ldsB[(n4 + 0) * LDBS + kk] = (__bf16)bR[i].x;
      ldsB[(n4 + 1) * LDBS + kk] = (__bf16)bR[i].y;
      ldsB[(n4 + 2) * LDBS + kk] = (__bf16)bR[i].z;
      ldsB[(n4 + 3) * LDBS + kk] = (__bf16)bR[i].w;
    }
  };

  async_a(0, 0);
  load_b(0);
  int p = 0;
  for (int k0 = 0; k0 < FDIM; k0 += 32) {
    store_b();
    wait_asynccnt0();   // this wave's A-chunk k0 writes to LDS are complete
    __syncthreads();    // everyone's A + B staged; prior compute finished
    if (k0 + 32 < FDIM) {
      async_a(k0 + 32, p ^ 1);  // fills alternate buffer during WMMAs
      load_b(k0 + 32);
    }

    const __bf16* aBuf = ldsA[p];
    v16bf a0 = frag_a(aBuf, waveM * 32 + 0, lane);
    v16bf a1 = frag_a(aBuf, waveM * 32 + 16, lane);
#pragma unroll
    for (int ns = 0; ns < 2; ++ns) {
      v16bf b = frag_b(ldsB, waveN * 32 + ns * 16, lane);
      acc[0][ns] = wmma_bf16(a0, b, acc[0][ns]);
      acc[1][ns] = wmma_bf16(a1, b, acc[1][ns]);
    }
    __syncthreads();
    p ^= 1;
  }

#pragma unroll
  for (int ms = 0; ms < 2; ++ms) {
#pragma unroll
    for (int ns = 0; ns < 2; ++ns) {
      int lmbase = waveM * 32 + ms * 16 + (lane >> 4) * 8;
      int n = n0 + waveN * 32 + ns * 16 + (lane & 15);
#pragma unroll
      for (int r = 0; r < 8; ++r) {
        int lm = lmbase + r;
        int m = mtile * 128 + lm;
        if (m < count) {
          float v = acc[ms][ns][r] * sLDS[lm];
          float* pdst = out + (size_t)tokLDS[lm] * DDIM + n;
          if (atomicMode) atomicAdd(pdst, v);
          else *pdst = v;
        }
      }
    }
  }
}

// ---------------------------------------------------------------------------
// Launch
// ---------------------------------------------------------------------------
extern "C" void kernel_launch(void* const* d_in, const int* in_sizes, int n_in,
                              void* d_out, int out_size, void* d_ws, size_t ws_size,
                              hipStream_t stream) {
  (void)in_sizes; (void)n_in; (void)out_size; (void)ws_size;
  const float* x   = (const float*)d_in[0];
  const float* gw  = (const float*)d_in[1];
  const float* w1  = (const float*)d_in[2];
  const float* w3  = (const float*)d_in[3];
  const float* w2  = (const float*)d_in[4];
  const float* sw1 = (const float*)d_in[5];
  const float* sw3 = (const float*)d_in[6];
  const float* sw2 = (const float*)d_in[7];
  const float* sgw = (const float*)d_in[8];
  float* out = (float*)d_out;

  // Workspace layout (~34.1 MB)
  char* ws = (char*)d_ws;
  float* cw = (float*)ws;          ws += (size_t)T_TOK * 8 * sizeof(float);
  float* sgate = (float*)ws;       ws += (size_t)T_TOK * sizeof(float);
  int* cnt = (int*)ws;             ws += 256;
  int* idx = (int*)ws;             ws += (size_t)NEXP * T_TOK * sizeof(int);
  __bf16* hbf = (__bf16*)ws;       // T_TOK * FDIM bf16 (32 MB), reused per expert

  (void)hipMemsetAsync(cnt, 0, NEXP * sizeof(int), stream);

  router_kernel<<<T_TOK / 8, 256, 0, stream>>>(x, gw, sgw, cw, sgate);
  scatter_kernel<<<T_TOK / 256, 256, 0, stream>>>(cw, cnt, idx);

  dim3 grid13(FDIM / 64, T_TOK / 128);  // (32, 64)
  dim3 grid2(DDIM / 64, T_TOK / 128);   // (16, 64)

  // Shared expert first: plain stores initialize out (harness poisons d_out).
  gemm13_kernel<<<grid13, 256, 0, stream>>>(x, sw1, sw3, nullptr, nullptr, T_TOK, hbf);
  gemm2_kernel<<<grid2, 256, 0, stream>>>(hbf, sw2, nullptr, nullptr, T_TOK,
                                          sgate, 1, out, /*atomic=*/0);

  // Routed experts: gathered rows, dynamic counts, atomic accumulation.
  for (int e = 0; e < NEXP; ++e) {
    gemm13_kernel<<<grid13, 256, 0, stream>>>(
        x, w1 + (size_t)e * DDIM * FDIM, w3 + (size_t)e * DDIM * FDIM,
        idx + e * T_TOK, cnt + e, 0, hbf);
    gemm2_kernel<<<grid2, 256, 0, stream>>>(
        hbf, w2 + (size_t)e * FDIM * DDIM, idx + e * T_TOK, cnt + e, 0,
        cw + e, 8, out, /*atomic=*/1);
  }
}
